// Attention_36876589203488
// MI455X (gfx1250) — matrix operand order repaired and checked by reading, not compiled, not run on hardware
//
#include <hip/hip_runtime.h>
#include <math.h>

// Problem constants (reference: B,H,S,D = 2,12,2048,64; fp32 everywhere)
#define B_ 2
#define H_ 12
#define S_ 2048
#define D_ 64

// log2-domain softmax: exp(x) == exp2(x*log2e); softmax invariant under the scaling.
#define SCALE_L2E 0.18033688011112042f   // (1/sqrt(64)) * log2(e)
#define NEG_INF_L2 (-1.442695e9f)        // -1e9 * log2(e)

#define QT 16            // query rows per workgroup (one WMMA M tile)
#define NKT (S_ / 16)    // 128 key tiles of 16
#define NWAVE 8
#define NTHREADS 256

// LDS score row stride padded to 2052 floats (8208 B == 16 B mod 256 B) so
// 16 lanes reading 16B chunks at row stride hit all 64 banks conflict-free.
#define SROW (S_ + 4)
#define SC_FLOATS  (QT * SROW)
#define OUT_FLOATS (QT * D_)
#define SMEM_FLOATS (SC_FLOATS + OUT_FLOATS + QT)
#define SMEM_BYTES (SMEM_FLOATS * 4)

typedef __attribute__((ext_vector_type(16))) _Float16 v16h;
typedef __attribute__((ext_vector_type(8)))  float    v8f;

union V16H { v16h v; _Float16 h[16]; };
union V8F  { v8f  v; float     f[8];  };

__global__ __launch_bounds__(NTHREADS)
void attn_wmma_kernel(const float* __restrict__ q,
                      const float* __restrict__ k,
                      const float* __restrict__ v,
                      const float* __restrict__ freq,
                      const int*   __restrict__ msk,
                      float* __restrict__ out,
                      float* __restrict__ pattn) {
  extern __shared__ float smem[];
  float* sc      = smem;                  // QT x SROW (padded) score/prob strip
  float* outacc  = smem + SC_FLOATS;      // QT x D_ accumulator
  float* rowstat = outacc + OUT_FLOATS;   // QT  (1/rowsum)

  const int tid   = threadIdx.x;
  const int wave  = tid >> 5;
  const int lane  = tid & 31;
  const int l16   = lane & 15;
  const int lhalf = lane >> 4;

  const int blk = blockIdx.x;
  const int qt  = blk % (S_ / QT);
  const int h   = (blk / (S_ / QT)) % H_;
  const int b   = blk / ((S_ / QT) * H_);

  const size_t bh    = (size_t)b * H_ + h;
  const float* qbase = q + (bh * S_ + (size_t)qt * QT) * D_;
  const float* kbase = k + bh * S_ * D_;
  const float* vbase = v + bh * S_ * D_;
  const size_t fm_base = ((size_t)b * S_ + (size_t)qt * QT) * S_; // freq/mask [B,1,S,S]

  for (int i = tid; i < OUT_FLOATS; i += NTHREADS) outacc[i] = 0.0f;

  // ---- Q tile -> WMMA A fragments (16x32 f16, contiguous k order) ----
  // lanes 0-15: M=lane, K = {0..15}; lanes 16-31: K = {16..31}
  const int abase = lhalf * 16;
  const float* qrow = qbase + (size_t)l16 * D_;
  V16H aq[2];
  #pragma unroll
  for (int step = 0; step < 2; ++step) {
    #pragma unroll
    for (int e = 0; e < 16; ++e) {
      const int kk = step * 32 + abase + e;
      aq[step].h[e] = (_Float16)qrow[kk];
    }
  }

  // ---- Phase 1: s' = (Q K^T)*scale*log2e + log2(freq); mask -> -inf'; into LDS ----
  for (int kt = wave; kt < NKT; kt += NWAVE) {
    const float* krow = kbase + (size_t)(kt * 16 + l16) * D_;
    if (kt + NWAVE < NKT)   // gfx1250 global_prefetch_b8 on next K tile
      __builtin_prefetch(kbase + (size_t)((kt + NWAVE) * 16 + l16) * D_, 0, 0);
    V8F c;
    #pragma unroll
    for (int i = 0; i < 8; ++i) c.f[i] = 0.0f;
    #pragma unroll
    for (int step = 0; step < 2; ++step) {
      // B fragment 32x16 f16: lane -> col n=l16; element e -> K = lhalf*16+e
      V16H bk;
      const int dbase = step * 32 + lhalf * 16;
      #pragma unroll
      for (int e = 0; e < 16; ++e) bk.h[e] = (_Float16)krow[dbase + e];
      c.v = __builtin_amdgcn_wmma_f32_16x16x32_f16(false, aq[step].v, false, bk.v,
                                                   (short)0, c.v, false, false);
    }
    const int kcol = kt * 16 + l16;
    #pragma unroll
    for (int r = 0; r < 8; ++r) {
      const int m = r + lhalf * 8;               // C layout: VGPR r -> row r / r+8
      const size_t fi = fm_base + (size_t)m * S_ + kcol;
      // raw v_log_f32 (log2); softmax is invariant to the log2e rescale
      float s = __builtin_fmaf(c.f[r], SCALE_L2E, __builtin_amdgcn_logf(freq[fi]));
      if (msk[fi] == 0) s = NEG_INF_L2;
      sc[m * SROW + kcol] = s;
    }
  }
  __syncthreads();

  // ---- Phase 2: row-wise softmax in exp2 domain, float4-vectorized ----
  #pragma unroll
  for (int rr = 0; rr < 2; ++rr) {
    const int row = wave * 2 + rr;
    float4* srow = (float4*)(sc + row * SROW);
    float mx = -3.0e38f;
    for (int j = lane; j < S_ / 4; j += 32) {
      const float4 t = srow[j];
      mx = fmaxf(mx, fmaxf(fmaxf(t.x, t.y), fmaxf(t.z, t.w)));
    }
    #pragma unroll
    for (int off = 16; off >= 1; off >>= 1) mx = fmaxf(mx, __shfl_xor(mx, off, 32));
    float sum = 0.0f;
    for (int j = lane; j < S_ / 4; j += 32) {
      float4 t = srow[j];
      t.x = __builtin_amdgcn_exp2f(t.x - mx);   // raw v_exp_f32
      t.y = __builtin_amdgcn_exp2f(t.y - mx);
      t.z = __builtin_amdgcn_exp2f(t.z - mx);
      t.w = __builtin_amdgcn_exp2f(t.w - mx);
      srow[j] = t;                              // keep unnormalized prob in LDS
      sum += (t.x + t.y) + (t.z + t.w);
    }
    #pragma unroll
    for (int off = 16; off >= 1; off >>= 1) sum += __shfl_xor(sum, off, 32);
    if (lane == 0) rowstat[row] = 1.0f / sum;
  }
  __syncthreads();

  // ---- Phase 2.5: stream normalized p_attn to HBM (b128 stores) ----
  {
    float4* pbase4 = (float4*)(pattn + bh * S_ * S_ + (size_t)qt * QT * S_);
    for (int i = tid; i < QT * (S_ / 4); i += NTHREADS) {
      const int row  = i >> 9;        // / (S_/4)
      const int col4 = i & (S_ / 4 - 1);
      float4 t = ((float4*)(sc + row * SROW))[col4];
      const float inv = rowstat[row];
      t.x *= inv; t.y *= inv; t.z *= inv; t.w *= inv;
      pbase4[i] = t;
    }
  }

  // ---- Phase 3: out = P @ V (unnormalized accumulate, scale at writeout) ----
  V8F c4[4];
  #pragma unroll
  for (int nt = 0; nt < 4; ++nt)
    #pragma unroll
    for (int i = 0; i < 8; ++i) c4[nt].f[i] = 0.0f;

  for (int ks = wave; ks < S_ / 32; ks += NWAVE) {
    const int kstart = ks * 32;
    // A fragment from LDS prob strip (same 16x32 f16 layout as Q); padded
    // stride makes these b128 DS reads bank-conflict-free.
    V16H ap;
    const float* prow = sc + l16 * SROW + kstart + abase;
    #pragma unroll
    for (int e = 0; e < 8; ++e) ap.h[e]     = (_Float16)prow[e];
    #pragma unroll
    for (int e = 0; e < 8; ++e) ap.h[8 + e] = (_Float16)prow[8 + e];
    // B fragments from V: element e -> K row kstart + lhalf*16 + e, col nt*16+l16
    const float* vrow = vbase + (size_t)(kstart + lhalf * 16) * D_ + l16;
    #pragma unroll
    for (int nt = 0; nt < 4; ++nt) {
      V16H bv;
      #pragma unroll
      for (int e = 0; e < 16; ++e) bv.h[e] = (_Float16)vrow[(size_t)e * D_ + nt * 16];
      c4[nt].v = __builtin_amdgcn_wmma_f32_16x16x32_f16(false, ap.v, false, bv.v,
                                                        (short)0, c4[nt].v, false, false);
    }
  }
  // Cross-wave merge via LDS float atomics (ds_add_f32)
  #pragma unroll
  for (int nt = 0; nt < 4; ++nt)
    #pragma unroll
    for (int r = 0; r < 8; ++r) {
      const int m = r + lhalf * 8;
      const int n = nt * 16 + l16;
      atomicAdd(&outacc[m * D_ + n], c4[nt].f[r]);
    }
  __syncthreads();

  // ---- Phase 4: scale by 1/rowsum, write out (b128 stores) ----
  {
    float4* obase4 = (float4*)(out + (bh * S_ + (size_t)qt * QT) * D_);
    for (int i = tid; i < QT * (D_ / 4); i += NTHREADS) {
      const int row = i >> 4;         // / (D_/4)
      float4 t = ((float4*)outacc)[i];
      const float inv = rowstat[row];
      t.x *= inv; t.y *= inv; t.z *= inv; t.w *= inv;
      obase4[i] = t;
    }
  }
}

extern "C" void kernel_launch(void* const* d_in, const int* in_sizes, int n_in,
                              void* d_out, int out_size, void* d_ws, size_t ws_size,
                              hipStream_t stream) {
  (void)in_sizes; (void)n_in; (void)d_ws; (void)ws_size; (void)out_size;

  const float* q    = (const float*)d_in[0];
  const float* k    = (const float*)d_in[1];
  const float* v    = (const float*)d_in[2];
  const float* freq = (const float*)d_in[3];
  const int*   msk  = (const int*)d_in[4];

  float* out   = (float*)d_out;                               // [B,H,S,D]
  float* pattn = out + (size_t)B_ * H_ * S_ * D_;             // [B,H,S,S]

  // Opt in to >64KB dynamic LDS (CDNA5 WGP: 320 KB). Idempotent, deterministic.
  (void)hipFuncSetAttribute((const void*)attn_wmma_kernel,
                            hipFuncAttributeMaxDynamicSharedMemorySize, SMEM_BYTES);

  const int grid = B_ * H_ * (S_ / QT);   // 3072 workgroups
  attn_wmma_kernel<<<grid, NTHREADS, SMEM_BYTES, stream>>>(q, k, v, freq, msk, out, pattn);
}
